// LightProtoSSM_29695403884551
// MI455X (gfx1250) — compile-verified, matrix operand order checked by reading, not run on hardware
//
#include <hip/hip_runtime.h>

// ---------------------------------------------------------------------------
// LightProtoSSM forward for MI455X (gfx1250, wave32, WMMA bf16).
// All GEMMs -> v_wmma_f32_16x16x32_bf16. fp32 storage, bf16 matrix inputs
// (native packed v_cvt via __builtin_convertvector), fp32 accumulation.
// Elementwise/LN/softmax/scan in fp32.
// ---------------------------------------------------------------------------

typedef __attribute__((ext_vector_type(16))) __bf16 v16bf;
typedef __attribute__((ext_vector_type(16))) float  v16f;
typedef __attribute__((ext_vector_type(8)))  float  v8f;

__device__ __forceinline__ v8f wmma_bf16(const v16bf& a, const v16bf& b, const v8f& c) {
  return __builtin_amdgcn_wmma_f32_16x16x32_bf16(
      /*neg_a=*/false, a, /*neg_b=*/false, b,
      /*c_mod=*/(short)0, c, /*reuse_a=*/false, /*reuse_b=*/false);
}

// ---------------------------------------------------------------------------
// Generic batched WMMA GEMM:  C = alpha * A @ op(B) + bias
//   A: (M x K) row-major, lda.  M % 128 == 0, K % 32 == 0 (true for all uses).
//   TRANSB=1: B is (N x K) row-major (weights / K-matrix), element B[n*ldb+k]
//   TRANSB=0: B is (K x N) row-major (V-matrix),           element B[k*ldb+n]
//   batch z: offsets = (z/inner)*sXo + (z%inner)*sXi
// Block = 256 threads = 8 waves (4 M x 2 N); each wave owns a 32x32 tile
// (2x2 of 16x16 WMMA tiles) -> block tile 128x64, 4 WMMAs per k-step.
// Fragment layouts per CDNA5 ISA 7.12.2 (wave32):
//   A lane l: row m = l&15, half h = l>>4; pair j, slot t:
//       k = 16*(j>>2) + 8*h + 2*(j&3) + t
//   B lane l: col n = l&15, same k permutation.
//   C lane l: col n = l&15, row m = v + 8*h for acc element v.
// ---------------------------------------------------------------------------
template <int TRANSB>
__global__ void __launch_bounds__(256) wmma_gemm_kernel(
    const float* __restrict__ A, const float* __restrict__ B,
    const float* __restrict__ bias, float* __restrict__ C,
    int M, int Nvalid, int K, int lda, int ldb, int ldc,
    int inner, long sAo, long sAi, long sBo, long sBi, long sCo, long sCi,
    float alpha)
{
  int bz = blockIdx.z;
  int bo = bz / inner, bi = bz - bo * inner;
  const float* Ab = A + bo * sAo + bi * sAi;
  const float* Bb = B + bo * sBo + bi * sBi;
  float*       Cb = C + bo * sCo + bi * sCi;

  int wave = threadIdx.x >> 5;
  int lane = threadIdx.x & 31;
  int mTile = blockIdx.x * 128 + (wave >> 1) * 32;
  int nTile = blockIdx.y * 64  + (wave & 1) * 32;
  if (mTile >= M || nTile >= Nvalid) return;   // wave-uniform: EXEC stays all-1s

  int h  = lane >> 4;
  int ml = lane & 15;
  const float* Arow0 = Ab + (size_t)(mTile + ml) * lda;
  const float* Arow1 = Arow0 + (size_t)16 * lda;

  int n0 = nTile + ml, n1 = n0 + 16;
  bool nOK0 = n0 < Nvalid, nOK1 = n1 < Nvalid;
  bool hasN1 = (nTile + 16) < Nvalid;          // wave-uniform
  int nS0 = nOK0 ? n0 : Nvalid - 1;
  int nS1 = nOK1 ? n1 : Nvalid - 1;
  float msk0 = nOK0 ? 1.f : 0.f;
  float msk1 = nOK1 ? 1.f : 0.f;

  const float* Brow0;
  const float* Brow1;
  if (TRANSB) { Brow0 = Bb + (size_t)nS0 * ldb; Brow1 = Bb + (size_t)nS1 * ldb; }
  else        { Brow0 = Bb + nS0;               Brow1 = Bb + nS1; }

  v8f a00 = {}, a01 = {}, a10 = {}, a11 = {};
#pragma unroll 2
  for (int kk = 0; kk < K; kk += 32) {
    v16f a0f, a1f, b0f, b1f;
#pragma unroll
    for (int j = 0; j < 8; ++j) {
      int ko = ((j >> 2) << 4) + (h << 3) + ((j & 3) << 1);
      float2 p0 = *(const float2*)(Arow0 + kk + ko);
      float2 p1 = *(const float2*)(Arow1 + kk + ko);
      a0f[2*j] = p0.x; a0f[2*j+1] = p0.y;
      a1f[2*j] = p1.x; a1f[2*j+1] = p1.y;
      if (TRANSB) {
        float2 q0 = *(const float2*)(Brow0 + kk + ko);
        b0f[2*j]   = q0.x * msk0;
        b0f[2*j+1] = q0.y * msk0;
        if (hasN1) {
          float2 q1 = *(const float2*)(Brow1 + kk + ko);
          b1f[2*j]   = q1.x * msk1;
          b1f[2*j+1] = q1.y * msk1;
        }
      } else {
        size_t r0 = (size_t)(kk + ko) * ldb;     // ko*ldb hoisted (unroll-const)
        b0f[2*j]   = Brow0[r0]       * msk0;
        b0f[2*j+1] = Brow0[r0 + ldb] * msk0;
        if (hasN1) {
          b1f[2*j]   = Brow1[r0]       * msk1;
          b1f[2*j+1] = Brow1[r0 + ldb] * msk1;
        }
      }
    }
    if (kk + 64 <= K) {                      // gfx1250 global_prefetch path
      __builtin_prefetch(Arow0 + kk + 32, 0, 1);
      __builtin_prefetch(Arow1 + kk + 32, 0, 1);
      if (TRANSB) {
        __builtin_prefetch(Brow0 + kk + 32, 0, 1);
        if (hasN1) __builtin_prefetch(Brow1 + kk + 32, 0, 1);
      }
    }
    // whole-vector fptrunc -> packed v_cvt bf16 (RNE)
    v16bf af0 = __builtin_convertvector(a0f, v16bf);
    v16bf af1 = __builtin_convertvector(a1f, v16bf);
    v16bf bf0 = __builtin_convertvector(b0f, v16bf);
    a00 = wmma_bf16(af0, bf0, a00);
    a10 = wmma_bf16(af1, bf0, a10);
    if (hasN1) {                             // wave-uniform branch
      v16bf bf1 = __builtin_convertvector(b1f, v16bf);
      a01 = wmma_bf16(af0, bf1, a01);
      a11 = wmma_bf16(af1, bf1, a11);
    }
  }

  float bb0 = (bias && nOK0) ? bias[n0] : 0.f;
  float bb1 = (bias && nOK1) ? bias[n1] : 0.f;
#pragma unroll
  for (int v = 0; v < 8; ++v) {
    int m0 = mTile + v + 8 * h;
    int m1 = m0 + 16;
    if (nOK0) {
      Cb[(size_t)m0 * ldc + n0] = alpha * a00[v] + bb0;
      Cb[(size_t)m1 * ldc + n0] = alpha * a10[v] + bb0;
    }
    if (nOK1) {
      Cb[(size_t)m0 * ldc + n1] = alpha * a01[v] + bb1;
      Cb[(size_t)m1 * ldc + n1] = alpha * a11[v] + bb1;
    }
  }
}

// ---------------------------------------------------------------------------
// meta[b, o] = meta_w[o,:] . concat(site_emb[sid[b]], hour_emb[hr[b]]) + meta_b[o]
// ---------------------------------------------------------------------------
__global__ void __launch_bounds__(256) kern_meta(
    const float* __restrict__ site_emb, const float* __restrict__ hour_emb,
    const int* __restrict__ sid, const int* __restrict__ hr,
    const float* __restrict__ mw, const float* __restrict__ mb,
    float* __restrict__ meta)
{
  int idx = blockIdx.x * 256 + threadIdx.x;   // B*512 threads
  int o = idx & 511; int b = idx >> 9;
  const float* se = site_emb + (size_t)sid[b] * 64;
  const float* he = hour_emb + (size_t)hr[b]  * 64;
  const float* wr = mw + (size_t)o * 128;
  float acc = mb[o];
#pragma unroll 8
  for (int k = 0; k < 64; ++k) acc += wr[k]      * se[k];
#pragma unroll 8
  for (int k = 0; k < 64; ++k) acc += wr[64 + k] * he[k];
  meta[idx] = acc;
}

// ---------------------------------------------------------------------------
// h = gelu(LN(h)) + pos_enc[t] + meta[b]        (in place, one block per row)
// ---------------------------------------------------------------------------
__global__ void __launch_bounds__(256) kern_post_input(
    float* __restrict__ h, const float* __restrict__ lw, const float* __restrict__ lb,
    const float* __restrict__ pos, const float* __restrict__ meta, int T)
{
  __shared__ float2 red[256];
  int row = blockIdx.x;            // b*T + t
  int b = row / T, t = row - b * T;
  float* hr = h + (size_t)row * 512;
  int i0 = threadIdx.x * 2;
  float x0 = hr[i0], x1 = hr[i0 + 1];
  red[threadIdx.x] = make_float2(x0 + x1, x0 * x0 + x1 * x1);
  __syncthreads();
  for (int s = 128; s > 0; s >>= 1) {
    if (threadIdx.x < s) {
      red[threadIdx.x].x += red[threadIdx.x + s].x;
      red[threadIdx.x].y += red[threadIdx.x + s].y;
    }
    __syncthreads();
  }
  float mu  = red[0].x * (1.f / 512.f);
  float var = red[0].y * (1.f / 512.f) - mu * mu;
  float rs  = rsqrtf(var + 1e-5f);
#pragma unroll
  for (int q = 0; q < 2; ++q) {
    int c = i0 + q;
    float y = ((q ? x1 : x0) - mu) * rs * lw[c] + lb[c];
    y = 0.5f * y * (1.f + erff(y * 0.70710678118654752f));     // exact GELU
    hr[c] = y + pos[(size_t)t * 512 + c] + meta[(size_t)b * 512 + c];
  }
}

// ---------------------------------------------------------------------------
// out = LN(a + r) * w + b     (one block per 512-elem row; in-place safe)
// ---------------------------------------------------------------------------
__global__ void __launch_bounds__(256) kern_add_ln(
    const float* __restrict__ a, const float* __restrict__ r,
    const float* __restrict__ w, const float* __restrict__ bb,
    float* __restrict__ out)
{
  __shared__ float2 red[256];
  int row = blockIdx.x;
  const float* ar = a + (size_t)row * 512;
  const float* rr = r + (size_t)row * 512;
  int i0 = threadIdx.x * 2;
  float x0 = ar[i0]     + rr[i0];
  float x1 = ar[i0 + 1] + rr[i0 + 1];
  red[threadIdx.x] = make_float2(x0 + x1, x0 * x0 + x1 * x1);
  __syncthreads();
  for (int s = 128; s > 0; s >>= 1) {
    if (threadIdx.x < s) {
      red[threadIdx.x].x += red[threadIdx.x + s].x;
      red[threadIdx.x].y += red[threadIdx.x + s].y;
    }
    __syncthreads();
  }
  float mu  = red[0].x * (1.f / 512.f);
  float var = red[0].y * (1.f / 512.f) - mu * mu;
  float rs  = rsqrtf(var + 1e-5f);
  float* orow = out + (size_t)row * 512;
  orow[i0]     = (x0 - mu) * rs * w[i0]     + bb[i0];
  orow[i0 + 1] = (x1 - mu) * rs * w[i0 + 1] + bb[i0 + 1];
}

// out[b,t,:] = in[b,T-1-t,:]
__global__ void __launch_bounds__(256) kern_reverse(
    const float* __restrict__ in, float* __restrict__ out, int T, int D, int total)
{
  int idx = blockIdx.x * 256 + threadIdx.x;
  if (idx >= total) return;
  int c = idx % D; int rw = idx / D; int t = rw % T; int b = rw / T;
  out[idx] = in[((size_t)(b * T + (T - 1 - t))) * D + c];
}

// xc = silu(causal depthwise conv4(xs) + conv_b)
__global__ void __launch_bounds__(256) kern_conv_silu(
    const float* __restrict__ xs, const float* __restrict__ cw,
    const float* __restrict__ cb, float* __restrict__ xc, int T, int D, int total)
{
  int idx = blockIdx.x * 256 + threadIdx.x;
  if (idx >= total) return;
  int c = idx % D; int rw = idx / D; int t = rw % T;
  float acc = cb[c];
#pragma unroll
  for (int j = 0; j < 4; ++j) {
    int ts = t - 3 + j;
    if (ts >= 0) acc += cw[c * 4 + j] * xs[((size_t)(rw - t + ts)) * D + c];
  }
  xc[idx] = acc / (1.f + __expf(-acc));   // silu
}

// ---------------------------------------------------------------------------
// Selective scan: one thread per (b, d) channel, N=16 state in registers.
// dt = softplus(dtpre); dA = exp(-exp(A_log)*dt); h = h*dA + x*dt*bm;
// y = sum(h*cm); out[b, t'(rev), outOff+d] = y + x*Dp[d]
// ---------------------------------------------------------------------------
__global__ void __launch_bounds__(256) kern_scan(
    const float* __restrict__ x, const float* __restrict__ dtpre,
    const float* __restrict__ bmat, const float* __restrict__ cmat,
    const float* __restrict__ A_log, const float* __restrict__ Dp,
    float* __restrict__ out, int T, int D, int outLd, int outOff, int revOut)
{
  int gid = blockIdx.x * 256 + threadIdx.x;   // B*D threads
  int d = gid % D; int b = gid / D;
  float an[16], hs[16];
#pragma unroll
  for (int n = 0; n < 16; ++n) { an[n] = -__expf(A_log[d * 16 + n]); hs[n] = 0.f; }
  float dpar = Dp[d];
  for (int t = 0; t < T; ++t) {
    size_t rw = (size_t)(b * T + t);
    float dtv = dtpre[rw * D + d];
    dtv = dtv > 20.f ? dtv : log1pf(__expf(dtv));
    float xv = x[rw * D + d];
    float xdt = xv * dtv;
    const float4* bp = (const float4*)(bmat + rw * 16);
    const float4* cp = (const float4*)(cmat + rw * 16);
    float y = 0.f;
#pragma unroll
    for (int q = 0; q < 4; ++q) {
      float4 bv = bp[q]; float4 cv = cp[q];
      int n = 4 * q;
      hs[n+0] = hs[n+0] * __expf(an[n+0] * dtv) + xdt * bv.x; y += hs[n+0] * cv.x;
      hs[n+1] = hs[n+1] * __expf(an[n+1] * dtv) + xdt * bv.y; y += hs[n+1] * cv.y;
      hs[n+2] = hs[n+2] * __expf(an[n+2] * dtv) + xdt * bv.z; y += hs[n+2] * cv.z;
      hs[n+3] = hs[n+3] * __expf(an[n+3] * dtv) + xdt * bv.w; y += hs[n+3] * cv.w;
    }
    int tt = revOut ? (T - 1 - t) : t;
    out[(size_t)(b * T + tt) * outLd + outOff + d] = y + xv * dpar;
  }
}

// Row softmax (256 cols), one wave32 per row, 8 rows per block.
__global__ void __launch_bounds__(256) kern_softmax(float* __restrict__ s, int rows)
{
  int wave = threadIdx.x >> 5, lane = threadIdx.x & 31;
  int row = blockIdx.x * 8 + wave;
  if (row >= rows) return;
  float* p = s + (size_t)row * 256;
  float v[8], mx = -1e30f;
#pragma unroll
  for (int i = 0; i < 8; ++i) { v[i] = p[lane + 32 * i]; mx = fmaxf(mx, v[i]); }
#pragma unroll
  for (int off = 16; off > 0; off >>= 1) mx = fmaxf(mx, __shfl_xor(mx, off, 32));
  float sum = 0.f;
#pragma unroll
  for (int i = 0; i < 8; ++i) { v[i] = __expf(v[i] - mx); sum += v[i]; }
#pragma unroll
  for (int off = 16; off > 0; off >>= 1) sum += __shfl_xor(sum, off, 32);
  float inv = 1.f / sum;
#pragma unroll
  for (int i = 0; i < 8; ++i) p[lane + 32 * i] = v[i] * inv;
}

// L2-normalize a 512-wide row: out = in / max(||in||, 1e-12)
__global__ void __launch_bounds__(256) kern_l2norm(
    const float* __restrict__ in, float* __restrict__ out)
{
  __shared__ float red[256];
  int row = blockIdx.x;
  const float* ir = in + (size_t)row * 512;
  int i0 = threadIdx.x * 2;
  float x0 = ir[i0], x1 = ir[i0 + 1];
  red[threadIdx.x] = x0 * x0 + x1 * x1;
  __syncthreads();
  for (int s = 128; s > 0; s >>= 1) {
    if (threadIdx.x < s) red[threadIdx.x] += red[threadIdx.x + s];
    __syncthreads();
  }
  float inv = 1.f / fmaxf(sqrtf(red[0]), 1e-12f);
  float* orow = out + (size_t)row * 512;
  orow[i0] = x0 * inv; orow[i0 + 1] = x1 * inv;
}

// out = sig(fa[c]) * (sim*softplus(temp) + cb[c]) + (1-sig(fa[c])) * perch
__global__ void __launch_bounds__(256) kern_final(
    const float* __restrict__ sim, const float* __restrict__ perch,
    const float* __restrict__ temp, const float* __restrict__ cbias,
    const float* __restrict__ fal, float* __restrict__ out, int total)
{
  int idx = blockIdx.x * 256 + threadIdx.x;
  if (idx >= total) return;
  int c = idx % 206;
  float tp = temp[0];
  tp = tp > 20.f ? tp : log1pf(__expf(tp));
  float a = 1.f / (1.f + __expf(-fal[c]));
  out[idx] = a * (sim[idx] * tp + cbias[c]) + (1.f - a) * perch[idx];
}

// ---------------------------------------------------------------------------
// Host side
// ---------------------------------------------------------------------------
static void gemm(hipStream_t stream,
                 const float* A, const float* Bm, const float* bias, float* Cm,
                 int M, int Nv, int K, int lda, int ldb, int ldc,
                 float alpha, int transB,
                 int batches = 1, int inner = 1,
                 long sAo = 0, long sAi = 0, long sBo = 0, long sBi = 0,
                 long sCo = 0, long sCi = 0)
{
  dim3 g((M + 127) / 128, (Nv + 63) / 64, batches);
  if (transB)
    wmma_gemm_kernel<1><<<g, 256, 0, stream>>>(A, Bm, bias, Cm, M, Nv, K, lda, ldb, ldc,
                                               inner, sAo, sAi, sBo, sBi, sCo, sCi, alpha);
  else
    wmma_gemm_kernel<0><<<g, 256, 0, stream>>>(A, Bm, bias, Cm, M, Nv, K, lda, ldb, ldc,
                                               inner, sAo, sAi, sBo, sBi, sCo, sCi, alpha);
}

extern "C" void kernel_launch(void* const* d_in, const int* in_sizes, int n_in,
                              void* d_out, int out_size, void* d_ws, size_t ws_size,
                              hipStream_t stream)
{
  (void)in_sizes; (void)n_in; (void)out_size; (void)ws_size;
  const int Bn = 32, T = 256, BT = Bn * T, Din = 1280, D = 512, H = 8, NC = 206;

  // Input index map: setup_inputs() dict insertion order, params flattened
  // recursively (layers: 28 tensors each starting at 13).
  auto F = [&](int i) { return (const float*)d_in[i]; };
  const int* site = (const int*)d_in[2];
  const int* hour = (const int*)d_in[3];

  float* w = (float*)d_ws;
  size_t off = 0;
  auto alloc = [&](size_t n) { float* p = w + off; off += n; return p; };
  float* bufH      = alloc((size_t)BT * D);
  float* bufRes    = alloc((size_t)BT * D);
  float* bufX      = alloc((size_t)BT * D);          // reversed input (bwd dir)
  float* bufXs     = alloc((size_t)BT * D);          // x_ssm
  float* bufXc     = alloc((size_t)BT * D);          // conv+silu
  float* bufDt     = alloc((size_t)BT * D);          // dt pre-softplus
  float* bufB16    = alloc((size_t)BT * 16);
  float* bufC16    = alloc((size_t)BT * 16);
  float* bufCat    = alloc((size_t)BT * 2 * D);      // [h_fwd | h_bwd]
  float* bufQKV    = alloc((size_t)BT * 3 * D);
  float* bufScores = alloc((size_t)Bn * H * T * T);
  float* bufAttn   = alloc((size_t)BT * D);
  float* bufMeta   = alloc((size_t)Bn * D);
  float* bufHn     = alloc((size_t)BT * D);
  float* bufPn     = alloc((size_t)NC * D);
  float* bufSim    = alloc((size_t)BT * NC);
  float* bufTmp    = alloc((size_t)BT * D);

  // meta projection (tiny)
  kern_meta<<<(Bn * D) / 256, 256, 0, stream>>>(F(9), F(10), site, hour, F(11), F(12), bufMeta);

  // h = emb @ inp_w^T + inp_b ; then gelu(LN(h)) + pos + meta
  gemm(stream, F(0), F(4), F(5), bufH, BT, D, Din, Din, Din, D, 1.f, 1);
  kern_post_input<<<BT, 256, 0, stream>>>(bufH, F(6), F(7), F(8), bufMeta, T);

  for (int l = 0; l < 2; ++l) {
    int pb = 13 + 28 * l;
    hipMemcpyAsync(bufRes, bufH, (size_t)BT * D * sizeof(float),
                   hipMemcpyDeviceToDevice, stream);

    for (int dir = 0; dir < 2; ++dir) {
      int sb = pb + (dir ? 9 : 0);                 // fwd / bwd SSM params
      const float* xin = bufH;
      if (dir) {
        kern_reverse<<<(BT * D) / 256, 256, 0, stream>>>(bufH, bufX, T, D, BT * D);
        xin = bufX;
      }
      // x_ssm = x @ in_w[:D]^T  (only first 512 rows of in_w are ever used)
      gemm(stream, xin, F(sb + 0), nullptr, bufXs, BT, D, D, D, D, D, 1.f, 1);
      kern_conv_silu<<<(BT * D) / 256, 256, 0, stream>>>(bufXs, F(sb + 1), F(sb + 2),
                                                         bufXc, T, D, BT * D);
      gemm(stream, bufXc, F(sb + 3), F(sb + 4), bufDt, BT, D, D, D, D, D, 1.f, 1);
      gemm(stream, bufXc, F(sb + 7), nullptr, bufB16, BT, 16, D, D, D, 16, 1.f, 1);
      gemm(stream, bufXc, F(sb + 8), nullptr, bufC16, BT, 16, D, D, D, 16, 1.f, 1);
      kern_scan<<<(Bn * D) / 256, 256, 0, stream>>>(xin, bufDt, bufB16, bufC16,
                                                    F(sb + 5), F(sb + 6),
                                                    bufCat, T, D, 2 * D,
                                                    dir ? D : 0, dir ? 1 : 0);
    }

    // merge + residual LN
    gemm(stream, bufCat, F(pb + 18), F(pb + 19), bufTmp, BT, D, 2 * D, 2 * D, 2 * D, D, 1.f, 1);
    kern_add_ln<<<BT, 256, 0, stream>>>(bufTmp, bufRes, F(pb + 20), F(pb + 21), bufH);

    // MHA
    gemm(stream, bufH, F(pb + 22), F(pb + 23), bufQKV, BT, 3 * D, D, D, D, 3 * D, 1.f, 1);
    // scores[b,h] = Q K^T / 8   (batched: z = b*8 + h)
    gemm(stream, bufQKV, bufQKV + D, nullptr, bufScores,
         T, T, 64, 3 * D, 3 * D, T, 0.125f, 1,
         Bn * H, H, (long)T * 3 * D, 64, (long)T * 3 * D, 64,
         (long)H * T * T, (long)T * T);
    kern_softmax<<<(Bn * H * T) / 8, 256, 0, stream>>>(bufScores, Bn * H * T);
    // O[b,h] = P V   (B non-transposed: K x N)
    gemm(stream, bufScores, bufQKV + 2 * D, nullptr, bufAttn,
         T, 64, T, T, 3 * D, D, 1.f, 0,
         Bn * H, H, (long)H * T * T, (long)T * T, (long)T * 3 * D, 64,
         (long)T * D, 64);
    gemm(stream, bufAttn, F(pb + 24), F(pb + 25), bufTmp, BT, D, D, D, D, D, 1.f, 1);
    kern_add_ln<<<BT, 256, 0, stream>>>(bufTmp, bufH, F(pb + 26), F(pb + 27), bufH);
  }

  // prototype head
  kern_l2norm<<<BT, 256, 0, stream>>>(bufH, bufHn);
  kern_l2norm<<<NC, 256, 0, stream>>>(F(69), bufPn);
  gemm(stream, bufHn, bufPn, nullptr, bufSim, BT, NC, D, D, D, NC, 1.f, 1);
  kern_final<<<(BT * NC + 255) / 256, 256, 0, stream>>>(bufSim, F(1), F(70), F(71), F(72),
                                                        (float*)d_out, BT * NC);
}
